// EnBiMambaLayer_77206332113514
// MI455X (gfx1250) — matrix operand order repaired and checked by reading, not compiled, not run on hardware
//
#include <hip/hip_runtime.h>
#include <hip/hip_bf16.h>

// ---------------------------------------------------------------------------
// Mamba layer for MI455X (gfx1250, wave32).
// Big GEMMs: v_wmma_f32_16x16x32_f16 with async-to-LDS B-tile staging
// (global_load_async_to_lds_b128 + s_wait_asynccnt), double buffered.
// ---------------------------------------------------------------------------

#define D_MODEL 768
#define D_INNER 1536
#define D_STATE 16
#define D_CONV  4
#define DT_RANK 48
#define BATCH   2
#define SEQ     2048
#define BL      (BATCH * SEQ)           // 4096 rows
#define XPROJ_N (DT_RANK + 2 * D_STATE) // 80

typedef __attribute__((ext_vector_type(16))) _Float16 v16h;
typedef __attribute__((ext_vector_type(8)))  _Float16 v8h;
typedef __attribute__((ext_vector_type(8)))  float    v8f;

// ---------------------------------------------------------------------------
// f32 -> f16 conversion (for WMMA operands)
// ---------------------------------------------------------------------------
__global__ void cvt_f32_to_f16_kernel(const float* __restrict__ in,
                                      _Float16* __restrict__ out, int n) {
  int i = blockIdx.x * blockDim.x + threadIdx.x;
  if (i < n) out[i] = (_Float16)in[i];
}

// ---------------------------------------------------------------------------
// WMMA GEMM: C[M,N] = A[M,K] * B[N,K]^T   (A,B f16 row-major, C f32 row-major)
// Block = 8 waves (256 threads), block tile 128(M) x 64(N).
// B tile (64 cols x 32 k, 4KB) staged in LDS via async loads, double buffered,
// shared by all 8 waves. Each wave: 16x64 output via 4 f32 accumulators.
// Requires: M % 128 == 0, N % 64 == 0, K % 32 == 0.
// ---------------------------------------------------------------------------
#define GEMM_WAVES 8
__global__ void __launch_bounds__(32 * GEMM_WAVES)
wmma_gemm_f16f32_kernel(const _Float16* __restrict__ Ah,
                        const _Float16* __restrict__ Bh,
                        float* __restrict__ C, int N, int K) {
  __shared__ alignas(128) _Float16 smemB[2][64 * 32];  // 2 x 4 KB

  const int tid  = threadIdx.x;
  const int wave = tid >> 5;
  const int lane = tid & 31;
  const int hi   = lane >> 4;   // 0: lanes 0-15, 1: lanes 16-31
  const int mr   = lane & 15;

  const int m0 = (blockIdx.y * GEMM_WAVES + wave) * 16;
  const int n0 = blockIdx.x * 64;

  // Async-copy mapping: thread -> (column c, 16-byte quarter q) of the B tile.
  const int c = tid >> 2;       // 0..63
  const int q = tid & 3;        // 0..3
  const uint32_t voff = (uint32_t)(((size_t)(n0 + c) * K + q * 8) * 2); // bytes
  const uint32_t lds_base = (uint32_t)(uintptr_t)(&smemB[0][0]);
  const uint32_t lds_dst0 = lds_base + (uint32_t)(c * 64 + q * 16);    // bytes

  // A fragment: row (m0+mr); lanes 0-15 hold K {0..15}, lanes 16-31 {16..31}
  const _Float16* __restrict__ arow = Ah + (size_t)(m0 + mr) * K + hi * 16;

  const int nk = K / 32;

  // Prologue: prefetch k-step 0 into buffer 0.
  {
    const void* sbase = (const void*)Bh;
    asm volatile("global_load_async_to_lds_b128 %0, %1, %2"
                 :: "v"(lds_dst0), "v"(voff), "s"(sbase) : "memory");
  }

  v8f acc0 = {}, acc1 = {}, acc2 = {}, acc3 = {};

  for (int i = 0; i < nk; ++i) {
    // My async issues (current buffer) complete, then block-wide visibility.
    asm volatile("s_wait_asynccnt 0x0" ::: "memory");
    __syncthreads();

    // Prefetch next k-step into the other buffer (uniform branch: EXEC stays full).
    if (i + 1 < nk) {
      const void* sbase = (const void*)(Bh + (size_t)(i + 1) * 32);
      uint32_t dst = lds_dst0 + (uint32_t)(((i + 1) & 1) * (64 * 32 * 2));
      asm volatile("global_load_async_to_lds_b128 %0, %1, %2"
                   :: "v"(dst), "v"(voff), "s"(sbase) : "memory");
    }

    const int k0 = i * 32;
    v8h alo = *(const v8h*)(arow + k0);
    v8h ahi = *(const v8h*)(arow + k0 + 8);
    v16h a = __builtin_shufflevector(alo, ahi, 0, 1, 2, 3, 4, 5, 6, 7,
                                     8, 9, 10, 11, 12, 13, 14, 15);

    // B fragment j: column (j*16 + mr); lanes 0-15 K {0..15}, lanes 16-31 K {16..31}.
    const _Float16* bt = &smemB[i & 1][0];
    v16h b0 = *(const v16h*)(bt + (0 * 16 + mr) * 32 + hi * 16);
    v16h b1 = *(const v16h*)(bt + (1 * 16 + mr) * 32 + hi * 16);
    v16h b2 = *(const v16h*)(bt + (2 * 16 + mr) * 32 + hi * 16);
    v16h b3 = *(const v16h*)(bt + (3 * 16 + mr) * 32 + hi * 16);

    acc0 = __builtin_amdgcn_wmma_f32_16x16x32_f16(false, a, false, b0, (short)0, acc0, false, false);
    acc1 = __builtin_amdgcn_wmma_f32_16x16x32_f16(false, a, false, b1, (short)0, acc1, false, false);
    acc2 = __builtin_amdgcn_wmma_f32_16x16x32_f16(false, a, false, b2, (short)0, acc2, false, false);
    acc3 = __builtin_amdgcn_wmma_f32_16x16x32_f16(false, a, false, b3, (short)0, acc3, false, false);
  }

  // C/D layout: VGPR r -> M = m0 + r + 8*hi, N = n0 + j*16 + mr
  const int mbase = m0 + hi * 8;
#pragma unroll
  for (int r = 0; r < 8; ++r) {
    float* cr = C + (size_t)(mbase + r) * N + n0 + mr;
    cr[0]  = acc0[r];
    cr[16] = acc1[r];
    cr[32] = acc2[r];
    cr[48] = acc3[r];
  }
}

// ---------------------------------------------------------------------------
// Depthwise causal conv (width 4) + bias + SiLU over x_in (first half of xz)
// xz:  [BL, 2*D_INNER] f32 ; xconv: [BL, D_INNER] f32
// ---------------------------------------------------------------------------
__global__ void conv_silu_kernel(const float* __restrict__ xz,
                                 const float* __restrict__ conv_w,
                                 const float* __restrict__ conv_b,
                                 float* __restrict__ xconv) {
  int idx = blockIdx.x * blockDim.x + threadIdx.x;
  if (idx >= BL * D_INNER) return;
  int r = idx / D_INNER;
  int d = idx - r * D_INNER;
  int l = r & (SEQ - 1);

  float acc = conv_b[d];
#pragma unroll
  for (int j = 0; j < D_CONV; ++j) {
    int ls = l - (D_CONV - 1) + j;
    if (ls >= 0) {
      acc += conv_w[d * D_CONV + j] * xz[(size_t)(r - (D_CONV - 1) + j) * (2 * D_INNER) + d];
    }
  }
  float sig = 1.0f / (1.0f + __expf(-acc));
  xconv[idx] = acc * sig;
}

// ---------------------------------------------------------------------------
// x_dbl[r, c] = sum_k xconv[r,k] * W_xproj[c,k]   (c < 80, K = 1536)
// ---------------------------------------------------------------------------
__global__ void xproj_kernel(const float* __restrict__ xconv,
                             const float* __restrict__ W_xproj,
                             float* __restrict__ x_dbl) {
  __shared__ float srow[D_INNER];
  const int r = blockIdx.x;
  const float* row = xconv + (size_t)r * D_INNER;
  for (int i = threadIdx.x; i < D_INNER; i += blockDim.x) srow[i] = row[i];
  __syncthreads();

  int c = threadIdx.x;
  if (c < XPROJ_N) {
    const float* w = W_xproj + (size_t)c * D_INNER;
    float acc = 0.f;
    for (int k = 0; k < D_INNER; ++k) acc += srow[k] * w[k];
    x_dbl[(size_t)r * XPROJ_N + c] = acc;
  }
}

// ---------------------------------------------------------------------------
// dt[r, d] = softplus( sum_j x_dbl[r, j] * W_dt[d, j] + b_dt[d] )   (K = 48)
// ---------------------------------------------------------------------------
__global__ void dt_kernel(const float* __restrict__ x_dbl,
                          const float* __restrict__ W_dt,
                          const float* __restrict__ b_dt,
                          float* __restrict__ dt) {
  __shared__ float sraw[DT_RANK];
  const int r = blockIdx.x;
  if (threadIdx.x < DT_RANK) sraw[threadIdx.x] = x_dbl[(size_t)r * XPROJ_N + threadIdx.x];
  __syncthreads();

  for (int d = threadIdx.x; d < D_INNER; d += blockDim.x) {
    const float* w = W_dt + (size_t)d * DT_RANK;
    float acc = b_dt[d];
#pragma unroll 8
    for (int j = 0; j < DT_RANK; ++j) acc += sraw[j] * w[j];
    float sp = (acc > 20.0f) ? acc : log1pf(__expf(acc));
    dt[(size_t)r * D_INNER + d] = sp;
  }
}

// ---------------------------------------------------------------------------
// Selective scan: one thread per (b, d) channel. 16-state recurrence in regs.
// Fuses + x*D, SiLU(z) gating, and f16 conversion for the output GEMM.
// ---------------------------------------------------------------------------
__global__ void scan_kernel(const float* __restrict__ dt,
                            const float* __restrict__ xconv,
                            const float* __restrict__ x_dbl,
                            const float* __restrict__ xz,
                            const float* __restrict__ A_log,
                            const float* __restrict__ Dparam,
                            _Float16* __restrict__ yh) {
  int t = blockIdx.x * blockDim.x + threadIdx.x;
  if (t >= BATCH * D_INNER) return;
  int b = t / D_INNER;
  int d = t - b * D_INNER;

  float A[D_STATE], h[D_STATE];
#pragma unroll
  for (int n = 0; n < D_STATE; ++n) {
    A[n] = -__expf(A_log[d * D_STATE + n]);
    h[n] = 0.0f;
  }
  const float Dd = Dparam[d];

  for (int l = 0; l < SEQ; ++l) {
    size_t r = (size_t)b * SEQ + l;
    float dtv = dt[r * D_INNER + d];
    float xv  = xconv[r * D_INNER + d];
    const float* xd = x_dbl + r * XPROJ_N;
    float du = dtv * xv;
    float y = 0.0f;
#pragma unroll
    for (int n = 0; n < D_STATE; ++n) {
      float dA = __expf(dtv * A[n]);
      h[n] = dA * h[n] + du * xd[DT_RANK + n];          // B_t[n]
      y += h[n] * xd[DT_RANK + D_STATE + n];            // C_t[n]
    }
    y += xv * Dd;
    float zv = xz[r * (2 * D_INNER) + D_INNER + d];
    float g = zv / (1.0f + __expf(-zv));
    yh[r * D_INNER + d] = (_Float16)(y * g);
  }
}

// ---------------------------------------------------------------------------
// out = x + LayerNorm(mamba_out) * w + b     (block per row of 768)
// ---------------------------------------------------------------------------
__global__ void ln_residual_kernel(const float* __restrict__ mo,
                                   const float* __restrict__ x,
                                   const float* __restrict__ lnw,
                                   const float* __restrict__ lnb,
                                   float* __restrict__ out) {
  __shared__ float s1[256];
  __shared__ float s2[256];
  const int r = blockIdx.x;
  const int t = threadIdx.x;
  const float* row = mo + (size_t)r * D_MODEL;

  float a0 = row[t], a1 = row[t + 256], a2 = row[t + 512];
  s1[t] = a0 + a1 + a2;
  s2[t] = a0 * a0 + a1 * a1 + a2 * a2;
  __syncthreads();
  for (int s = 128; s > 0; s >>= 1) {
    if (t < s) { s1[t] += s1[t + s]; s2[t] += s2[t + s]; }
    __syncthreads();
  }
  float mu  = s1[0] * (1.0f / D_MODEL);
  float var = s2[0] * (1.0f / D_MODEL) - mu * mu;
  float inv = rsqrtf(var + 1e-6f);

#pragma unroll
  for (int j = 0; j < 3; ++j) {
    int i = t + j * 256;
    out[(size_t)r * D_MODEL + i] =
        x[(size_t)r * D_MODEL + i] + (row[i] - mu) * inv * lnw[i] + lnb[i];
  }
}

// ---------------------------------------------------------------------------
// Launch
// ---------------------------------------------------------------------------
extern "C" void kernel_launch(void* const* d_in, const int* in_sizes, int n_in,
                              void* d_out, int out_size, void* d_ws, size_t ws_size,
                              hipStream_t stream) {
  const float* x       = (const float*)d_in[0];   // [2,2048,768]
  const float* W_in    = (const float*)d_in[1];   // [3072,768]
  const float* conv_w  = (const float*)d_in[2];   // [1536,1,4]
  const float* conv_b  = (const float*)d_in[3];   // [1536]
  const float* W_xproj = (const float*)d_in[4];   // [80,1536]
  const float* W_dt    = (const float*)d_in[5];   // [1536,48]
  const float* b_dt    = (const float*)d_in[6];   // [1536]
  const float* A_log   = (const float*)d_in[7];   // [1536,16]
  const float* Dp      = (const float*)d_in[8];   // [1536]
  const float* W_out   = (const float*)d_in[9];   // [768,1536]
  const float* ln_w    = (const float*)d_in[10];  // [768]
  const float* ln_b    = (const float*)d_in[11];  // [768]
  float* out = (float*)d_out;

  char* ws = (char*)d_ws;
  size_t off = 0;
  _Float16* xh    = (_Float16*)(ws + off); off += (size_t)BL * D_MODEL * 2;          // 6.29 MB
  _Float16* Winh  = (_Float16*)(ws + off); off += (size_t)2 * D_INNER * D_MODEL * 2; // 4.72 MB
  _Float16* Wouth = (_Float16*)(ws + off); off += (size_t)D_MODEL * D_INNER * 2;     // 2.36 MB
  float* xz     = (float*)(ws + off); off += (size_t)BL * 2 * D_INNER * 4;           // 50.3 MB
  float* xconv  = (float*)(ws + off); off += (size_t)BL * D_INNER * 4;               // 25.2 MB
  float* x_dbl  = (float*)(ws + off); off += (size_t)BL * XPROJ_N * 4;               // 1.31 MB
  float* dt     = (float*)(ws + off); off += (size_t)BL * D_INNER * 4;               // 25.2 MB
  _Float16* yh  = (_Float16*)(ws + off); off += (size_t)BL * D_INNER * 2;            // 12.6 MB
  float* mo     = (float*)(ws + off); off += (size_t)BL * D_MODEL * 4;               // 12.6 MB
  (void)off; (void)ws_size; (void)in_sizes; (void)n_in; (void)out_size;

  // 1) f32 -> f16 operands
  {
    int n = BL * D_MODEL;
    cvt_f32_to_f16_kernel<<<(n + 255) / 256, 256, 0, stream>>>(x, xh, n);
    n = 2 * D_INNER * D_MODEL;
    cvt_f32_to_f16_kernel<<<(n + 255) / 256, 256, 0, stream>>>(W_in, Winh, n);
    n = D_MODEL * D_INNER;
    cvt_f32_to_f16_kernel<<<(n + 255) / 256, 256, 0, stream>>>(W_out, Wouth, n);
  }

  // 2) xz = x @ W_in^T : M=4096, N=3072, K=768
  {
    dim3 grid((2 * D_INNER) / 64, BL / (16 * GEMM_WAVES));
    wmma_gemm_f16f32_kernel<<<grid, 32 * GEMM_WAVES, 0, stream>>>(
        xh, Winh, xz, 2 * D_INNER, D_MODEL);
  }

  // 3) depthwise causal conv + SiLU
  {
    int n = BL * D_INNER;
    conv_silu_kernel<<<(n + 255) / 256, 256, 0, stream>>>(xz, conv_w, conv_b, xconv);
  }

  // 4) x_dbl = xconv @ W_xproj^T
  xproj_kernel<<<BL, 128, 0, stream>>>(xconv, W_xproj, x_dbl);

  // 5) dt = softplus(dt_raw @ W_dt^T + b_dt)
  dt_kernel<<<BL, 256, 0, stream>>>(x_dbl, W_dt, b_dt, dt);

  // 6) selective scan + gating -> yh (f16)
  {
    int n = BATCH * D_INNER;
    scan_kernel<<<(n + 127) / 128, 128, 0, stream>>>(dt, xconv, x_dbl, xz, A_log, Dp, yh);
  }

  // 7) mamba_out = y @ W_out^T : M=4096, N=768, K=1536
  {
    dim3 grid(D_MODEL / 64, BL / (16 * GEMM_WAVES));
    wmma_gemm_f16f32_kernel<<<grid, 32 * GEMM_WAVES, 0, stream>>>(
        yh, Wouth, mo, D_MODEL, D_INNER);
  }

  // 8) residual + LayerNorm
  ln_residual_kernel<<<BL, 256, 0, stream>>>(mo, x, ln_w, ln_b, out);
}
